// TritonLiteMLA_12859132084795
// MI455X (gfx1250) — matrix operand order repaired and checked by reading, not compiled, not run on hardware
//
#include <hip/hip_runtime.h>
#include <hip/hip_bf16.h>

typedef __bf16 bf16_t;
typedef __bf16 v16bf __attribute__((ext_vector_type(16)));
typedef __bf16 v8bf  __attribute__((ext_vector_type(8)));
typedef __bf16 v4bf  __attribute__((ext_vector_type(4)));
typedef float  v8f   __attribute__((ext_vector_type(8)));
typedef unsigned int u32;
typedef u32 u32x4 __attribute__((ext_vector_type(4)));
typedef u32 u32x8 __attribute__((ext_vector_type(8)));

#define NUM_HEADS 32
#define B_SZ 8
#define N_SEQ 4096
#define C_DIM 1024
#define QKV_O 3072
#define LDSTRIDE 40   // bf16 per LDS row: 80B = 20 dwords; 16B-aligned, conflict-free

__device__ __forceinline__ v8f zero_v8f() {
    v8f z = {0.f, 0.f, 0.f, 0.f, 0.f, 0.f, 0.f, 0.f};
    return z;
}

// A-fragment (16-bit A 16x32): K = e + half*16 (contiguous, matches B).
// Two 16B-aligned ds_load_b128 from a plain [row][k] tile.
__device__ __forceinline__ v16bf load_frag_a(const bf16_t* row, int half) {
    const v8bf lo = *(const v8bf*)(row + half * 16);
    const v8bf hi = *(const v8bf*)(row + half * 16 + 8);
    return __builtin_shufflevector(lo, hi, 0, 1, 2, 3, 4, 5, 6, 7,
                                   8, 9, 10, 11, 12, 13, 14, 15);
}
// B-fragment (16-bit B 32x16): K = e + half*16 -> contiguous 32B at [col][half*16].
__device__ __forceinline__ v16bf load_frag_b(const bf16_t* row, int half) {
    const v8bf lo = *(const v8bf*)(row + half * 16);
    const v8bf hi = *(const v8bf*)(row + half * 16 + 8);
    return __builtin_shufflevector(lo, hi, 0, 1, 2, 3, 4, 5, 6, 7,
                                   8, 9, 10, 11, 12, 13, 14, 15);
}

__device__ __forceinline__ v4bf cvt4(float4 f) {
    v4bf r = {(bf16_t)f.x, (bf16_t)f.y, (bf16_t)f.z, (bf16_t)f.w};
    return r;
}

// Flat LDS pointer -> LDS byte offset (low 32 bits of the flat address).
__device__ __forceinline__ u32 lds_off(const void* p) {
    return (u32)(uintptr_t)p;
}

// ---------------------------------------------------------------------------
// TDM: DMA a 2D bf16 tile (tile_w x tile_h, tile_w = 32 elems = 64B rows) from
// global (row stride row_stride_elems) into LDS, hardware-padding each 16-dword
// row with 4 dwords -> 80B LDS row stride (= LDSTRIDE).  Issued per-wave;
// completion tracked with TENSORcnt.  Descriptor per CDNA5 D# (groups 0/1, 2D).
// ---------------------------------------------------------------------------
__device__ __forceinline__ void tdm_load_tile_bf16(
        u32 lds_byte_off, const bf16_t* gptr,
        u32 tensor_w, u32 tensor_h, u32 row_stride_elems,
        u32 tile_w, u32 tile_h) {
    const unsigned long long ga = (unsigned long long)(uintptr_t)gptr;
    u32x4 g0;
    g0[0] = 1u;                                            // count=1, user mode
    g0[1] = lds_byte_off;                                  // lds_addr
    g0[2] = (u32)(ga & 0xFFFFFFFFu);                       // global_addr[31:0]
    g0[3] = (u32)((ga >> 32) & 0x01FFFFFFu) | (2u << 30);  // addr[56:32] | type=2
    u32x8 g1;
    g1[0] = (1u << 16)      // data_size = 2B
          | (1u << 20)      // pad_enable
          | (3u << 22)      // pad_interval: every 16 dwords (64B)
          | (3u << 25);     // pad_amount: 4 dwords (16B)
    g1[1] = (tensor_w & 0xFFFFu) << 16;                    // tensor_dim0[15:0]
    g1[2] = (tensor_w >> 16) | ((tensor_h & 0xFFFFu) << 16);
    g1[3] = (tensor_h >> 16) | (tile_w << 16);             // | tile_dim0
    g1[4] = tile_h & 0xFFFFu;                              // tile_dim1 (dim2=0)
    g1[5] = row_stride_elems;                              // dim0_stride[31:0]
    g1[6] = 0u;
    g1[7] = 0u;
    asm volatile("tensor_load_to_lds %0, %1" :: "s"(g0), "s"(g1) : "memory");
}

// ---------------------------------------------------------------------------
// Kernels 1 & 4: out[M,N] = act(A[M,K] @ W[N,K]^T) (+bias).
// 128x128x32 tiles; 8 waves; each wave 32x64 via 2x4 wmma_f32_16x16x32_bf16.
// TDM_A: A is bf16 and staged by the Tensor Data Mover; else f32 + cvt staging.
// ---------------------------------------------------------------------------
template <bool TDM_A, bool RELU, bool BIAS, bool OUT_BF16>
__global__ __launch_bounds__(256)
void gemm_wmma_kernel(const void* __restrict__ Ain,
                      const float* __restrict__ W,
                      const float* __restrict__ bias,
                      void* __restrict__ outp,
                      int M, int N, int K) {
    __shared__ __align__(16) bf16_t lds_a[128][LDSTRIDE];  // [m][k]
    __shared__ __align__(16) bf16_t lds_b[128][LDSTRIDE];  // [n][k]

    const int tid  = threadIdx.x;
    const int m0   = blockIdx.y * 128;
    const int n0   = blockIdx.x * 128;
    const int wid  = tid >> 5;
    const int wm   = wid >> 1;
    const int wn   = wid & 1;
    const int lane = tid & 31;
    const int r15  = lane & 15;
    const int half = lane >> 4;
    const u32 lds_a_byte = lds_off(&lds_a[0][0]);

    v8f acc[2][4];
#pragma unroll
    for (int mf = 0; mf < 2; ++mf)
#pragma unroll
        for (int nf = 0; nf < 4; ++nf) acc[mf][nf] = zero_v8f();

    for (int k0 = 0; k0 < K; k0 += 32) {
        if constexpr (TDM_A) {
            if (wid == 0)   // one wave drives the DMA of the whole A slab
                tdm_load_tile_bf16(lds_a_byte,
                                   (const bf16_t*)Ain + (size_t)m0 * K + k0,
                                   (u32)K, (u32)M, (u32)K, 32u, 128u);
        } else {
            const float* A = (const float*)Ain;
#pragma unroll
            for (int p = 0; p < 4; ++p) {
                const int idx = p * 256 + tid;
                const int r   = idx >> 3;
                const int c4  = (idx & 7) << 2;
                const float4 av = *(const float4*)(A + (size_t)(m0 + r) * K + k0 + c4);
                *(v4bf*)(&lds_a[r][c4]) = cvt4(av);
            }
            if (k0 + 32 < K)
                __builtin_prefetch(A + (size_t)(m0 + (tid >> 1)) * K + k0 + 32, 0, 3);
        }
#pragma unroll
        for (int p = 0; p < 4; ++p) {
            const int idx = p * 256 + tid;
            const int r   = idx >> 3;
            const int c4  = (idx & 7) << 2;
            const float4 wv = *(const float4*)(W + (size_t)(n0 + r) * K + k0 + c4);
            *(v4bf*)(&lds_b[r][c4]) = cvt4(wv);
        }
        if (k0 + 32 < K)
            __builtin_prefetch(W + (size_t)(n0 + (tid >> 1)) * K + k0 + 32, 0, 3);
        if constexpr (TDM_A) {
            if (wid == 0) __builtin_amdgcn_s_wait_tensorcnt(0);
        }
        __syncthreads();

        v16bf af[2];
#pragma unroll
        for (int mf = 0; mf < 2; ++mf)
            af[mf] = load_frag_a(&lds_a[wm * 32 + mf * 16 + r15][0], half);
        v16bf bf[4];
#pragma unroll
        for (int nf = 0; nf < 4; ++nf)
            bf[nf] = load_frag_b(&lds_b[wn * 64 + nf * 16 + r15][0], half);

#pragma unroll
        for (int mf = 0; mf < 2; ++mf)
#pragma unroll
            for (int nf = 0; nf < 4; ++nf)
                acc[mf][nf] = __builtin_amdgcn_wmma_f32_16x16x32_bf16(
                    false, af[mf], false, bf[nf], (short)0, acc[mf][nf], false, false);
        __syncthreads();
    }

#pragma unroll
    for (int mf = 0; mf < 2; ++mf)
#pragma unroll
        for (int nf = 0; nf < 4; ++nf)
#pragma unroll
            for (int i = 0; i < 8; ++i) {
                const int gm = m0 + wm * 32 + mf * 16 + i + half * 8;
                const int gn = n0 + wn * 64 + nf * 16 + r15;
                float v = acc[mf][nf][i];
                if (RELU) v = v > 0.f ? v : 0.f;
                if (BIAS) v += bias[gn];
                const size_t idx = (size_t)gm * N + gn;
                if constexpr (OUT_BF16) ((bf16_t*)outp)[idx] = (bf16_t)v;
                else                    ((float*)outp)[idx]  = v;
            }
}

// ---------------------------------------------------------------------------
// Kernel 2: vk[b,h,e,d] = sum_n v_pad[b,n,h,e] * k[b,n,h,d].
// One block (2 waves) per (b,h).  A = v^T [e][n], B = k^T [d][n] (transposed
// element stores).  Ones row (e=32) accumulated in registers during staging.
// ---------------------------------------------------------------------------
__global__ __launch_bounds__(64)
void vk_accum_kernel(const bf16_t* __restrict__ qkv, float* __restrict__ vkout) {
    __shared__ __align__(16) bf16_t lds_vT[32][LDSTRIDE];  // [e][n]
    __shared__ __align__(16) bf16_t lds_kT[32][LDSTRIDE];  // [d][n]
    __shared__ float part[8][32];

    const int tid  = threadIdx.x;       // 64
    const int bh   = blockIdx.x;
    const int b    = bh >> 5, h = bh & 31;
    const int wave = tid >> 5;
    const int lane = tid & 31;
    const int r15  = lane & 15;
    const int half = lane >> 4;

    v8f acc0 = zero_v8f(), acc1 = zero_v8f();
    float kacc[4] = {0.f, 0.f, 0.f, 0.f};
    const size_t base = (size_t)b * N_SEQ * QKV_O + h * 32;

    for (int n0 = 0; n0 < N_SEQ; n0 += 32) {
#pragma unroll
        for (int p = 0; p < 4; ++p) {
            const int t4 = p * 64 + tid;          // 0..255
            const int n  = t4 >> 3;               // 0..31
            const int c4 = (t4 & 7) << 2;
            const size_t g = base + (size_t)(n0 + n) * QKV_O;
            const v4bf vv = *(const v4bf*)(qkv + g + 2 * C_DIM + c4);
            const v4bf kv = *(const v4bf*)(qkv + g + 1 * C_DIM + c4);
#pragma unroll
            for (int j = 0; j < 4; ++j) {
                lds_vT[c4 + j][n] = vv[j];        // transpose (A K-dim = n)
                lds_kT[c4 + j][n] = kv[j];        // transpose (B K-dim = n)
                kacc[j] += (float)kv[j];          // ones-row partial
            }
        }
        __syncthreads();

        const v16bf af  = load_frag_a(&lds_vT[wave * 16 + r15][0], half);
        const v16bf bf0 = load_frag_b(&lds_kT[r15][0], half);
        const v16bf bf1 = load_frag_b(&lds_kT[16 + r15][0], half);
        acc0 = __builtin_amdgcn_wmma_f32_16x16x32_bf16(false, af, false, bf0,
                                                       (short)0, acc0, false, false);
        acc1 = __builtin_amdgcn_wmma_f32_16x16x32_bf16(false, af, false, bf1,
                                                       (short)0, acc1, false, false);
        __syncthreads();
    }

    float* dst = vkout + (size_t)bh * (33 * 32);
#pragma unroll
    for (int i = 0; i < 8; ++i) {
        const int e = wave * 16 + i + half * 8;
        dst[e * 32 + r15]      = acc0[i];
        dst[e * 32 + 16 + r15] = acc1[i];
    }
#pragma unroll
    for (int j = 0; j < 4; ++j) part[tid >> 3][((tid & 7) << 2) + j] = kacc[j];
    __syncthreads();
    if (tid < 32) {
        float s = 0.f;
#pragma unroll
        for (int g = 0; g < 8; ++g) s += part[g][tid];
        dst[32 * 32 + tid] = s;
    }
}

// ---------------------------------------------------------------------------
// Kernel 3: vk_q[n,e] = q[n,:] @ vk[e,:]^T per (b,h); divide by (den+eps);
// write proj_input (bf16).  q tile staged by TDM; denominator via a 3rd WMMA
// against vk's ones row (live col 0), broadcast with __shfl (wave32).
// ---------------------------------------------------------------------------
__global__ __launch_bounds__(256)
void vkq_div_kernel(const bf16_t* __restrict__ qkv, const float* __restrict__ vkin,
                    bf16_t* __restrict__ proj) {
    __shared__ __align__(16) bf16_t lds_q[128][LDSTRIDE];  // [n][d]

    const int tid  = threadIdx.x;
    const int bid  = blockIdx.x;
    const int bh   = bid >> 5;
    const int tile = bid & 31;
    const int b    = bh >> 5, h = bh & 31;
    const int n_base = tile * 128;
    const int wid  = tid >> 5;
    const int lane = tid & 31;
    const int r15  = lane & 15;
    const int half = lane >> 4;
    const float* vkp = vkin + (size_t)bh * (33 * 32);

    if (wid == 0) {   // DMA the whole 128x32 q tile (stride 3072) into LDS
        tdm_load_tile_bf16(lds_off(&lds_q[0][0]),
                           qkv + ((size_t)b * N_SEQ + n_base) * QKV_O + h * 32,
                           (u32)QKV_O, (u32)(B_SZ * N_SEQ), (u32)QKV_O, 32u, 128u);
        __builtin_amdgcn_s_wait_tensorcnt(0);
    }
    __syncthreads();

    // B fragments: B[d][e] = vk[e*32+d] (f32 -> bf16), d = e + half*16
    v16bf bf0, bf1, bf2;
    const float* p0 = vkp + r15 * 32 + half * 16;
    const float* p1 = vkp + (16 + r15) * 32 + half * 16;
    const float* pd = vkp + 32 * 32 + half * 16;          // ones row (den)
#pragma unroll
    for (int e = 0; e < 16; ++e) {
        bf0[e] = (bf16_t)p0[e];
        bf1[e] = (bf16_t)p1[e];
        bf2[e] = (r15 == 0) ? (bf16_t)pd[e] : (bf16_t)0.f;  // only col 0 live
    }

    const v16bf af = load_frag_a(&lds_q[wid * 16 + r15][0], half);

    v8f acc0 = zero_v8f(), acc1 = zero_v8f(), acc2 = zero_v8f();
    acc0 = __builtin_amdgcn_wmma_f32_16x16x32_bf16(false, af, false, bf0,
                                                   (short)0, acc0, false, false);
    acc1 = __builtin_amdgcn_wmma_f32_16x16x32_bf16(false, af, false, bf1,
                                                   (short)0, acc1, false, false);
    acc2 = __builtin_amdgcn_wmma_f32_16x16x32_bf16(false, af, false, bf2,
                                                   (short)0, acc2, false, false);

#pragma unroll
    for (int i = 0; i < 8; ++i) {
        const int ri   = i + half * 8;
        const float dn = __shfl(acc2[i], half * 16, 32) + 1e-15f;
        const int n    = n_base + wid * 16 + ri;
        const size_t o = ((size_t)b * N_SEQ + n) * C_DIM + h * 32;
        proj[o + r15]      = (bf16_t)(acc0[i] / dn);
        proj[o + 16 + r15] = (bf16_t)(acc1[i] / dn);
    }
}

// ---------------------------------------------------------------------------
extern "C" void kernel_launch(void* const* d_in, const int* in_sizes, int n_in,
                              void* d_out, int out_size, void* d_ws, size_t ws_size,
                              hipStream_t stream) {
    const float* x     = (const float*)d_in[0];
    const float* wqkv  = (const float*)d_in[1];
    const float* wproj = (const float*)d_in[2];
    const float* bproj = (const float*)d_in[3];
    float* y = (float*)d_out;

    const size_t QKV_BYTES = (size_t)B_SZ * N_SEQ * QKV_O * sizeof(bf16_t);
    const size_t VK_BYTES  = (size_t)B_SZ * NUM_HEADS * 33 * 32 * sizeof(float);
    bf16_t* qkv_ws  = (bf16_t*)d_ws;
    float*  vk_ws   = (float*)((char*)d_ws + QKV_BYTES);
    bf16_t* proj_ws = (bf16_t*)((char*)d_ws + QKV_BYTES + VK_BYTES);

    const int M = B_SZ * N_SEQ;  // 32768

    gemm_wmma_kernel<false, true, false, true>
        <<<dim3(QKV_O / 128, M / 128), 256, 0, stream>>>(x, wqkv, nullptr, qkv_ws,
                                                         M, QKV_O, C_DIM);
    vk_accum_kernel<<<B_SZ * NUM_HEADS, 64, 0, stream>>>(qkv_ws, vk_ws);
    vkq_div_kernel<<<B_SZ * NUM_HEADS * (N_SEQ / 128), 256, 0, stream>>>(qkv_ws, vk_ws,
                                                                         proj_ws);
    gemm_wmma_kernel<true, false, true, false>
        <<<dim3(C_DIM / 128, M / 128), 256, 0, stream>>>(proj_ws, wproj, bproj, y,
                                                         M, C_DIM, C_DIM);
}
